// GetSpatialGatingWeights_12996571038073
// MI455X (gfx1250) — matrix operand order repaired and checked by reading, not compiled, not run on hardware
//
#include <hip/hip_runtime.h>
#include <math.h>

typedef unsigned short u16;
typedef __attribute__((ext_vector_type(16))) __bf16 v16bf;
typedef __attribute__((ext_vector_type(8)))  float  v8f;
typedef __attribute__((ext_vector_type(8)))  unsigned short v8u;
typedef __attribute__((ext_vector_type(16))) unsigned short v16u;
typedef __attribute__((ext_vector_type(4)))  unsigned int   v4ui;
typedef __attribute__((ext_vector_type(8)))  int            v8i;
typedef __attribute__((ext_vector_type(4)))  int            v4i;

#define NN 4
#define PER_N (64L*32L*64L*64L)

#if defined(__has_builtin)
#if __has_builtin(__builtin_amdgcn_tensor_load_to_lds)
#define HAVE_TDM 1
#endif
#endif
#if __has_include(<hip/amd_detail/amd_gfx1250_TDM.h>)
#define TDM_6ARG 1
#endif

__device__ __forceinline__ u16 f2bf(float f) {
    unsigned int u = __float_as_uint(f);
    unsigned int r = (u + 0x7FFFu + ((u >> 16) & 1u)) >> 16;
    return (u16)r;
}

__device__ __forceinline__ v16bf ld_frag(const u16* p0, const u16* p1) {
    v8u lo = *(const v8u*)p0;
    v8u hi = *(const v8u*)p1;
    v16u t = __builtin_shufflevector(lo, hi, 0,1,2,3,4,5,6,7,8,9,10,11,12,13,14,15);
    return __builtin_bit_cast(v16bf, t);
}

__device__ __forceinline__ v8f wmma_bf16(v16bf a, v16bf b, v8f c) {
    return __builtin_amdgcn_wmma_f32_16x16x32_bf16(false, a, false, b, (short)0, c, false, false);
}

__device__ __forceinline__ unsigned lds_off(const void* p) {
    // generic pointer to LDS: low 32 bits carry the LDS byte address (ISA 10.2)
    return (unsigned)(unsigned long long)p;
}

__device__ __forceinline__ void wait_tensor0() {
#if defined(HAVE_TDM)
#if __has_builtin(__builtin_amdgcn_s_wait_tensorcnt)
    __builtin_amdgcn_s_wait_tensorcnt(0);
#else
    asm volatile("s_wait_tensorcnt 0x0" ::: "memory");
#endif
#endif
}

#if defined(HAVE_TDM)
// 2-D bf16 tile load via Tensor Data Mover. tile_d0 elems per row (dim0),
// tile_d1 rows; row stride in elements; optional LDS padding (codes per ISA:
// pad_interval 3 -> every 16 DWORDs, pad_amount 3 -> 4 DWORDs). pad_amt<0 = off.
__device__ __forceinline__ void tdm_load_2d(unsigned ldsaddr, const void* gptr,
                                            unsigned tile_d0, unsigned tile_d1,
                                            unsigned long long stride_elems,
                                            int pad_int_code, int pad_amt_code) {
    unsigned long long ga = (unsigned long long)gptr;
    v4ui g0;
    g0.x = 1u;                                               // count=1 (valid, user)
    g0.y = ldsaddr;                                          // lds_addr [63:32]
    g0.z = (unsigned)(ga & 0xFFFFFFFFu);                     // global_addr lo
    g0.w = (unsigned)((ga >> 32) & 0x01FFFFFFu) | (2u << 30);// addr[56:32] | type=2
    unsigned d0 = (1u << 16);                                // data_size=1 (2 bytes)
    if (pad_amt_code >= 0)
        d0 |= (1u << 20) | ((unsigned)pad_int_code << 22) | ((unsigned)pad_amt_code << 25);
    unsigned td0 = tile_d0, td1 = tile_d1;                   // tensor dims == tile dims
    v8i g1;
    g1[0] = (int)d0;
    g1[1] = (int)((td0 & 0xFFFFu) << 16);                    // abar=0 | tensor_dim0 lo
    g1[2] = (int)(((td1 & 0xFFFFu) << 16) | (td0 >> 16));    // tensor_dim0 hi | dim1 lo
    g1[3] = (int)(((tile_d0 & 0xFFFFu) << 16) | (td1 >> 16));// dim1 hi | tile_dim0
    g1[4] = (int)(tile_d1 & 0xFFFFu);                        // tile_dim1 | tile_dim2=0
    g1[5] = (int)(unsigned)(stride_elems & 0xFFFFFFFFull);   // dim0_stride lo
    g1[6] = (int)(unsigned)((stride_elems >> 32) & 0xFFFFull); // stride hi | dim1_stride=0
    g1[7] = 0;
    v4i z4 = {0, 0, 0, 0};
#if defined(TDM_6ARG)
    v8i z8 = {0, 0, 0, 0, 0, 0, 0, 0};
    __builtin_amdgcn_tensor_load_to_lds(g0, g1, z4, z4, z8, 0);
#else
    __builtin_amdgcn_tensor_load_to_lds(g0, g1, z4, z4, 0);
#endif
}
#endif

// ---------------- stats ----------------
__global__ void k_zero_stats(float* stats) {
    if (threadIdx.x < 16) stats[threadIdx.x] = 0.0f;
}

__global__ void k_reduce(const float* __restrict__ x, float* __restrict__ stats) {
    const int n = blockIdx.y;
    const float4* px = (const float4*)(x + (long)n * PER_N);
    const long n4 = PER_N / 4;
    float s = 0.0f, ss = 0.0f;
    for (long i = (long)blockIdx.x * blockDim.x + threadIdx.x; i < n4;
         i += (long)gridDim.x * blockDim.x) {
        float4 v = px[i];
        s  += v.x + v.y + v.z + v.w;
        ss += v.x*v.x + v.y*v.y + v.z*v.z + v.w*v.w;
    }
    __shared__ float sm[256], sm2[256];
    int t = threadIdx.x;
    sm[t] = s; sm2[t] = ss;
    __syncthreads();
    for (int o = 128; o > 0; o >>= 1) {
        if (t < o) { sm[t] += sm[t + o]; sm2[t] += sm2[t + o]; }
        __syncthreads();
    }
    if (t == 0) {
        atomicAdd(&stats[n], sm[0]);
        atomicAdd(&stats[4 + n], sm2[0]);
    }
}

__global__ void k_finalize(float* stats) {
    int t = threadIdx.x;
    if (t < NN) {
        float cnt = (float)PER_N;
        float mu  = stats[t] / cnt;
        float var = stats[4 + t] / cnt - mu * mu;
        stats[8 + t]  = mu;
        stats[12 + t] = rsqrtf(var + 1e-5f);
    }
}

// ---------------- weight conversion ----------------
__global__ void k_cvt(const float* __restrict__ W1, const float* __restrict__ W2,
                      const float* __restrict__ W3, const float* __restrict__ W4,
                      u16* __restrict__ w1b, u16* __restrict__ w2b,
                      u16* __restrict__ w3t, u16* __restrict__ w4b) {
    int i = blockIdx.x * blockDim.x + threadIdx.x;
    if (i < 128 * 64)  w1b[i] = f2bf(W1[i]);
    if (i < 512 * 512) w2b[i] = f2bf(W2[i]);
    if (i < 256 * 256) {               // transpose: w3t[p][k] = W3[k][p]
        int k = i >> 8, p = i & 255;
        w3t[p * 256 + k] = f2bf(W3[i]);
    }
    if (i < 64 * 128)  w4b[i] = f2bf(W4[i]);
}

// ---------------- stage 1: norm + W1 + gelu + block-scatter ----------------
// grid: (h=64, d=32, n=4), 256 threads (8 waves)
__global__ void k_stage1(const float* __restrict__ x, const float* __restrict__ b1,
                         const u16* __restrict__ W1bf, const float* __restrict__ stats,
                         u16* __restrict__ Ub, u16* __restrict__ Vb) {
    const int h = blockIdx.x, d = blockIdx.y, n = blockIdx.z;
    __shared__ u16 Wl[128 * 64];     // W1 bf16, row-major [k][c]
    __shared__ u16 xt[64 * 72];      // normalized x, [w][c] (pitch 72)
    const int t = threadIdx.x;
    const int lane = t & 31, wid = t >> 5;

#if defined(HAVE_TDM)
    if (wid == 0)
        tdm_load_2d(lds_off(Wl), W1bf, 64, 128, 64ull, -1, -1);  // contiguous 16KB
#else
    for (int i = t; i < 128 * 64 / 8; i += 256)
        *(v8u*)&Wl[i * 8] = *(const v8u*)&W1bf[i * 8];
#endif

    const float mu = stats[8 + n], rs = stats[12 + n];
    {
        int c  = t >> 2;
        int w0 = (t & 3) * 16;
        const float* px = x + ((((long)(n * 64 + c) * 32 + d) * 64 + h) * 64 + w0);
        #pragma unroll
        for (int j = 0; j < 16; j += 4) {
            float4 v = *(const float4*)(px + j);
            xt[(w0 + j + 0) * 72 + c] = f2bf((v.x - mu) * rs);
            xt[(w0 + j + 1) * 72 + c] = f2bf((v.y - mu) * rs);
            xt[(w0 + j + 2) * 72 + c] = f2bf((v.z - mu) * rs);
            xt[(w0 + j + 3) * 72 + c] = f2bf((v.w - mu) * rs);
        }
    }
#if defined(HAVE_TDM)
    if (wid == 0) wait_tensor0();
#endif
    __syncthreads();

    const int m0 = wid * 16;
    const int l15 = lane & 15, half = lane >> 4;
    v8f acc[4] = {};
    #pragma unroll
    for (int ks = 0; ks < 64; ks += 32) {
        const u16* ra = &Wl[(m0 + l15) * 64 + ks + 8 * half];
        v16bf afr = ld_frag(ra, ra + 16);
        #pragma unroll
        for (int nt = 0; nt < 4; nt++) {
            const u16* rb = &xt[(nt * 16 + l15) * 72 + ks + 8 * half];
            v16bf bfr = ld_frag(rb, rb + 16);
            acc[nt] = wmma_bf16(afr, bfr, acc[nt]);
        }
    }
    #pragma unroll
    for (int nt = 0; nt < 4; nt++) {
        int w = nt * 16 + l15;
        int g = ((d >> 2) << 6) + ((h >> 3) << 3) + (w >> 3);
        int p = ((d & 3) << 6) + ((h & 7) << 3) + (w & 7);
        #pragma unroll
        for (int r = 0; r < 8; r++) {
            int kk = m0 + r + 8 * half;
            float val = acc[nt][r] + b1[kk];
            val = 0.5f * val * (1.0f + erff(val * 0.70710678118654752f));
            int cc = kk & 63;
            u16* dst = (kk < 64) ? Ub : Vb;
            dst[(((long)(n * 64 + cc) * 512 + g) << 8) + p] = f2bf(val);
        }
    }
}

// ---------------- generic batched bf16 WMMA GEMM ----------------
// C[m,n] = sum_k A[m,k]*B[k,n] + bias ; A (MxK) rm, B (KxN) rm, C (MxN) rm bf16
// grid: (N/128, M/128, batch), 256 threads (8 waves), 128x128 tile
// A tile streamed by TDM (pad -> pitch 40); B tile transposed manually.
__global__ void k_gemm(const u16* __restrict__ A, long sA,
                       const u16* __restrict__ B, long sB,
                       u16* __restrict__ C, long sC,
                       int M, int N, int K,
                       const float* __restrict__ bias, int biasMode) {
    const int b  = blockIdx.z;
    const u16* Ab = A + (long)b * sA;
    const u16* Bb = B + (long)b * sB;
    u16* Cb = C + (long)b * sC;
    const int m0 = blockIdx.y * 128, n0 = blockIdx.x * 128;

    __shared__ u16 As[128 * 40];   // [m][k] pitch 40 (= 32 + 8 pad -> TDM pad 4 DW / 16 DW)
    __shared__ u16 Bt[128 * 40];   // [n][k] pitch 40 (transposed stage)

    const int t = threadIdx.x, lane = t & 31, wid = t >> 5;
    const int wm = (wid & 3) * 32, wn = (wid >> 2) * 64;
    const int l15 = lane & 15, half = lane >> 4;
    v8f acc[2][4] = {};

    for (int kk = 0; kk < K; kk += 32) {
#if defined(HAVE_TDM)
        if (wid == 0)   // DMA 128x32 bf16 tile, rows padded 16B to pitch-40
            tdm_load_2d(lds_off(As), Ab + (long)m0 * K + kk,
                        32, 128, (unsigned long long)K, 3, 3);
#else
        {   // stage A tile 128x32
            int i = t >> 1, j0 = (t & 1) * 16;
            const u16* src = Ab + (long)(m0 + i) * K + kk + j0;
            v8u a0 = *(const v8u*)src;
            v8u a1 = *(const v8u*)(src + 8);
            *(v8u*)&As[i * 40 + j0]     = a0;
            *(v8u*)&As[i * 40 + j0 + 8] = a1;
        }
#endif
        {   // stage B tile 32x128, transposed into Bt[n][k]
            int i = t >> 3, j0 = (t & 7) * 16;
            const u16* src = Bb + (long)(kk + i) * N + n0 + j0;
            if (kk + 32 < K)   // prefetch 32 elements ahead in the same B row (global_prefetch_b8)
                __builtin_prefetch(src + 32, 0, 1);
            v8u b0 = *(const v8u*)src;
            v8u b1v = *(const v8u*)(src + 8);
            #pragma unroll
            for (int e = 0; e < 8; e++) Bt[(j0 + e) * 40 + i]     = b0[e];
            #pragma unroll
            for (int e = 0; e < 8; e++) Bt[(j0 + 8 + e) * 40 + i] = b1v[e];
        }
#if defined(HAVE_TDM)
        if (wid == 0) wait_tensor0();
#endif
        __syncthreads();

        v16bf afr[2], bfr[4];
        #pragma unroll
        for (int im = 0; im < 2; im++) {
            const u16* ra = &As[(wm + im * 16 + l15) * 40 + 8 * half];
            afr[im] = ld_frag(ra, ra + 16);
        }
        #pragma unroll
        for (int in = 0; in < 4; in++) {
            const u16* rb = &Bt[(wn + in * 16 + l15) * 40 + 8 * half];
            bfr[in] = ld_frag(rb, rb + 16);
        }
        #pragma unroll
        for (int im = 0; im < 2; im++)
            #pragma unroll
            for (int in = 0; in < 4; in++)
                acc[im][in] = wmma_bf16(afr[im], bfr[in], acc[im][in]);
        __syncthreads();
    }

    #pragma unroll
    for (int im = 0; im < 2; im++)
        #pragma unroll
        for (int in = 0; in < 4; in++) {
            int gn = n0 + wn + in * 16 + l15;
            #pragma unroll
            for (int r = 0; r < 8; r++) {
                int gm = m0 + wm + im * 16 + r + 8 * half;
                float v = acc[im][in][r] +
                          ((biasMode == 0) ? bias[gm] : bias[gn]);
                Cb[(long)gm * N + gn] = f2bf(v);
            }
        }
}

// ---------------- stage 3: concat + W4 + unblock ----------------
// grid: (g=512, p-half=2, n=4), 256 threads (8 waves)
__global__ void k_out(const u16* __restrict__ UM, const u16* __restrict__ VM,
                      const u16* __restrict__ W4bf, const float* __restrict__ b4,
                      float* __restrict__ out) {
    const int g = blockIdx.x, ph = blockIdx.y, n = blockIdx.z;
    const int p0 = ph * 128;
    __shared__ u16 Tt[128 * 136];   // [p-local][c] pitch 136
    __shared__ u16 Wl[64 * 128];    // W4 bf16 [j][c]
    const int t = threadIdx.x, lane = t & 31, wid = t >> 5;

#if defined(HAVE_TDM)
    if (wid == 0)
        tdm_load_2d(lds_off(Wl), W4bf, 128, 64, 128ull, -1, -1); // contiguous 16KB
#else
    for (int i = t; i < 64 * 128 / 8; i += 256)
        *(v8u*)&Wl[i * 8] = *(const v8u*)&W4bf[i * 8];
#endif

    #pragma unroll
    for (int it = 0; it < 8; it++) {
        int idx = t + it * 256;       // 0..2047 (v8 chunks)
        int c  = idx >> 4;            // 0..127
        int j0 = (idx & 15) * 8;      // 0..120
        const u16* src = (c < 64)
            ? UM + ((((long)(n * 64 + c)       * 512 + g) << 8) + p0 + j0)
            : VM + ((((long)(n * 64 + (c-64))  * 512 + g) << 8) + p0 + j0);
        v8u v = *(const v8u*)src;
        #pragma unroll
        for (int e = 0; e < 8; e++) Tt[(j0 + e) * 136 + c] = v[e];
    }
#if defined(HAVE_TDM)
    if (wid == 0) wait_tensor0();
#endif
    __syncthreads();

    const int l15 = lane & 15, half = lane >> 4;
    const int wm = (wid & 3) * 16;
    const int wn = (wid >> 2) * 64;
    v8f acc[4] = {};
    #pragma unroll
    for (int kk = 0; kk < 128; kk += 32) {
        const u16* ra = &Wl[(wm + l15) * 128 + kk + 8 * half];
        v16bf afr = ld_frag(ra, ra + 16);
        #pragma unroll
        for (int in = 0; in < 4; in++) {
            const u16* rb = &Tt[(wn + in * 16 + l15) * 136 + kk + 8 * half];
            v16bf bfr = ld_frag(rb, rb + 16);
            acc[in] = wmma_bf16(afr, bfr, acc[in]);
        }
    }
    #pragma unroll
    for (int in = 0; in < 4; in++) {
        int p  = p0 + wn + in * 16 + l15;
        int d  = ((g >> 6) << 2) + (p >> 6);
        int hh = (((g >> 3) & 7) << 3) + ((p >> 3) & 7);
        int ww = ((g & 7) << 3) + (p & 7);
        #pragma unroll
        for (int r = 0; r < 8; r++) {
            int j = wm + r + 8 * half;
            float v = acc[in][r] + b4[j];
            out[(((long)(n * 64 + j) * 32 + d) * 64 + hh) * 64 + ww] = v;
        }
    }
}

extern "C" void kernel_launch(void* const* d_in, const int* in_sizes, int n_in,
                              void* d_out, int out_size, void* d_ws, size_t ws_size,
                              hipStream_t stream) {
    const float* x  = (const float*)d_in[0];
    const float* W1 = (const float*)d_in[1];
    const float* b1 = (const float*)d_in[2];
    const float* W2 = (const float*)d_in[3];
    const float* b2 = (const float*)d_in[4];
    const float* W3 = (const float*)d_in[5];
    const float* b3 = (const float*)d_in[6];
    const float* W4 = (const float*)d_in[7];
    const float* b4 = (const float*)d_in[8];

    char* ws = (char*)d_ws;
    size_t o = 0;
    float* stats = (float*)(ws + o); o += 256;
    u16* w1b = (u16*)(ws + o); o += 128 * 64 * 2;
    u16* w2b = (u16*)(ws + o); o += 512 * 512 * 2;
    u16* w3t = (u16*)(ws + o); o += 256 * 256 * 2;
    u16* w4b = (u16*)(ws + o); o += 64 * 128 * 2;
    o = (o + 255) & ~(size_t)255;
    const size_t half_bytes = (size_t)4 * 64 * 512 * 256 * 2;   // 67 MB each
    u16* Ub = (u16*)(ws + o); o += half_bytes;
    u16* Vb = (u16*)(ws + o); o += half_bytes;
    u16* UM = (u16*)(ws + o); o += half_bytes;
    u16* VM = (u16*)(ws + o); o += half_bytes;

    k_zero_stats<<<1, 32, 0, stream>>>(stats);
    k_reduce<<<dim3(512, 4), 256, 0, stream>>>(x, stats);
    k_finalize<<<1, 32, 0, stream>>>(stats);
    k_cvt<<<1024, 256, 0, stream>>>(W1, W2, W3, W4, w1b, w2b, w3t, w4b);
    k_stage1<<<dim3(64, 32, 4), 256, 0, stream>>>(x, b1, w1b, stats, Ub, Vb);
    // u-mix: Out_u[k,p] = W2[k,g] * Ub[g,p] + b2[k]   (A shared, bias per-row)
    k_gemm<<<dim3(2, 4, 256), 256, 0, stream>>>(w2b, 0, Ub, 512 * 256, UM, 512 * 256,
                                                512, 256, 512, b2, 0);
    // v-mix: Out_v[g,k] = Vb[g,p] * W3t[p,k] + b3[k]  (B shared, bias per-col)
    k_gemm<<<dim3(2, 4, 256), 256, 0, stream>>>(Vb, 512 * 256, w3t, 0, VM, 512 * 256,
                                                512, 256, 256, b3, 1);
    k_out<<<dim3(512, 2, 4), 256, 0, stream>>>(UM, VM, w4b, b4, (float*)d_out);
}
